// StandardAttentionModule_87342454931676
// MI455X (gfx1250) — matrix operand order repaired and checked by reading, not compiled, not run on hardware
//
#include <hip/hip_runtime.h>

// ---------------------------------------------------------------------------
// Transposed-channel attention block for MI455X (gfx1250, wave32, WMMA).
// All heavy math via v_wmma_f32_16x16x32_bf16 (f32 accum), softmax/norms f32.
// Tile staging uses CDNA5 async global->LDS copies + prefetch.
// ---------------------------------------------------------------------------

#define DIMC   384
#define HEADS  8
#define HD     48
#define BATCH  4
#define HH     96
#define WW     96
#define NS     (HH * WW)     // 9216 spatial
#define C3     (3 * DIMC)    // 1152

typedef __attribute__((ext_vector_type(16))) __bf16 v16bf;
typedef __attribute__((ext_vector_type(8)))  __bf16 v8bf;
typedef __attribute__((ext_vector_type(8)))  float  v8f;

__device__ __forceinline__ __bf16 f2bf(float f) {
  unsigned u = __float_as_uint(f);
  u = (u + 0x7FFFu + ((u >> 16) & 1u)) >> 16;           // round-to-nearest-even
  unsigned short h = (unsigned short)u;
  __bf16 b; __builtin_memcpy(&b, &h, 2); return b;
}
__device__ __forceinline__ float bf2f(__bf16 b) {
  unsigned short h; __builtin_memcpy(&h, &b, 2);
  return __uint_as_float(((unsigned)h) << 16);
}
__device__ __forceinline__ v16bf mk16(v8bf lo, v8bf hi) {
  v16bf r;
#pragma unroll
  for (int i = 0; i < 8; ++i) { r[i] = lo[i]; r[i + 8] = hi[i]; }
  return r;
}
#define WMMA_BF16(a, b, c) \
  __builtin_amdgcn_wmma_f32_16x16x32_bf16(false, (a), false, (b), (short)0, (c), false, false)

// CDNA5 async global->LDS copy (ASYNCcnt), 16B granule. LDS dest is the
// wave-relative offset = low 32 bits of the flat shared-memory address.
__device__ __forceinline__ unsigned lds_off_of(const void* p) {
  return (unsigned)(unsigned long long)(uintptr_t)p;
}
__device__ __forceinline__ void async_copy_b128(unsigned lds_off, const void* gptr) {
  asm volatile("global_load_async_to_lds_b128 %0, %1, off"
               :: "v"(lds_off), "v"(gptr) : "memory");
}
__device__ __forceinline__ void wait_async0() {
  asm volatile("s_wait_asynccnt 0x0" ::: "memory");
}

// ---------------------------------------------------------------------------
// f32 -> bf16 elementwise convert
// ---------------------------------------------------------------------------
__global__ void cvt_bf16_kernel(const float* __restrict__ src, __bf16* __restrict__ dst, int n) {
  int i = blockIdx.x * 256 + threadIdx.x;
  if (i < n) dst[i] = f2bf(src[i]);
}

// enh_w [co][ci][3][3] f32 -> [tap][co][ci] bf16 (contiguous per-tap GEMM A)
__global__ void repack_enh_kernel(const float* __restrict__ w, __bf16* __restrict__ dst) {
  int i = blockIdx.x * 256 + threadIdx.x;
  const int total = C3 * C3 * 9;
  if (i >= total) return;
  int tap = i % 9;
  int rc  = i / 9;                                      // co*C3 + ci
  dst[(long)tap * C3 * C3 + rc] = f2bf(w[i]);
}

// ---------------------------------------------------------------------------
// Generic bf16 GEMM: D[M,N] = A[M,K] * B[K,N], row-major, K % 32 == 0.
// Block 256 thr = 8 waves; tile BM=BN=128, BK=32; wave = 4x2 WMMA 16x16 tiles.
// Per-z operand offsets: z -> (zb = z/zmod, zh = z%zmod), off = zb*s1 + zh*s2.
// A tile staged with async global->LDS; B tile transposed at store time.
// ---------------------------------------------------------------------------
template <bool STORE_BF16>
__global__ __launch_bounds__(256)
void gemm_kernel(const __bf16* __restrict__ A, const __bf16* __restrict__ B,
                 void* __restrict__ Dv, int M, int N, int K, int zmod,
                 long sA1, long sA2, long sB1, long sB2, long sD1, long sD2) {
  __shared__ __bf16 sA[128][40];   // row stride 80B keeps 16B-aligned chunks
  __shared__ __bf16 sB[128][40];   // stored transposed: sB[n][k]
  int z  = blockIdx.z;
  int zb = z / zmod, zh = z % zmod;
  A += zb * sA1 + zh * sA2;
  B += zb * sB1 + zh * sB2;
  long doff = zb * sD1 + zh * sD2;

  int n0 = blockIdx.x * 128, m0 = blockIdx.y * 128;
  int tid  = threadIdx.x;
  int lane = tid & 31, wid = tid >> 5;
  int half = lane >> 4, l16 = lane & 15;
  int wm = (wid >> 2) * 64, wn = (wid & 3) * 32;

  v8f acc[4][2] = {};
  int arow = tid >> 1, acb = (tid & 1) * 16;   // A: 2 thr/row, 16 bf16 each
  int bk = tid >> 3,  bn  = (tid & 7) * 16;    // B: 8 thr/k-row, 16 n each
  unsigned ldsA0 = lds_off_of(&sA[arow][acb]);
  unsigned ldsA1 = lds_off_of(&sA[arow][acb + 8]);

  for (int k0 = 0; k0 < K; k0 += 32) {
    __syncthreads();
    int gm = m0 + arow;
    { // A tile via async copy (zero-fill rows >= M)
      if (gm < M) {
        const __bf16* s = A + (long)gm * K + k0 + acb;
        async_copy_b128(ldsA0, s);
        async_copy_b128(ldsA1, s + 8);
      } else {
#pragma unroll
        for (int i = 0; i < 16; ++i) sA[arow][acb + i] = (__bf16)0.0f;
      }
    }
    { // B tile, transposed into LDS
      const __bf16* s = B + (long)(k0 + bk) * N + n0 + bn;
      v8bf a0 = *(const v8bf*)(s);
      v8bf a1 = *(const v8bf*)(s + 8);
#pragma unroll
      for (int i = 0; i < 8; ++i) { sB[bn + i][bk] = a0[i]; sB[bn + 8 + i][bk] = a1[i]; }
    }
    if (k0 + 32 < K) {  // prefetch next K tile (in-bounds by construction)
      if (gm < M) __builtin_prefetch(A + (long)gm * K + k0 + 32 + acb, 0, 1);
      __builtin_prefetch(B + (long)(k0 + 32 + bk) * N + n0 + bn, 0, 1);
    }
    wait_async0();
    __syncthreads();

    v16bf af[4], bfv[2];
#pragma unroll
    for (int i = 0; i < 4; ++i) {           // A frag: K = half*8+e | 16+half*8+e
      const __bf16* p = &sA[wm + i * 16 + l16][0];
      af[i] = mk16(*(const v8bf*)(p + half * 8), *(const v8bf*)(p + 16 + half * 8));
    }
#pragma unroll
    for (int j = 0; j < 2; ++j) {           // B frag: K = half*8+e | 16+half*8+e
      const __bf16* p = &sB[wn + j * 16 + l16][0];
      bfv[j] = mk16(*(const v8bf*)(p + half * 8), *(const v8bf*)(p + 16 + half * 8));
    }
#pragma unroll
    for (int i = 0; i < 4; ++i)
#pragma unroll
      for (int j = 0; j < 2; ++j)
        acc[i][j] = WMMA_BF16(af[i], bfv[j], acc[i][j]);
  }

  // D layout: VGPR v -> m = v + half*8 ; n = l16
#pragma unroll
  for (int i = 0; i < 4; ++i)
#pragma unroll
    for (int j = 0; j < 2; ++j)
#pragma unroll
      for (int v = 0; v < 8; ++v) {
        int m = m0 + wm + i * 16 + half * 8 + v;
        int n = n0 + wn + j * 16 + l16;
        if (m < M) {
          if (STORE_BF16) ((__bf16*)Dv)[doff + (long)m * N + n] = f2bf(acc[i][j][v]);
          else            ((float*)Dv)[doff + (long)m * N + n]  = acc[i][j][v];
        }
      }
}

// ---------------------------------------------------------------------------
// 3x3 conv as 9-tap implicit GEMM: out[co][p] = sum_tap sum_ci Wt[tap][co][ci]*in_sh[ci][p]
// Same WMMA core; B tile applies the spatial shift with zero padding.
// ---------------------------------------------------------------------------
__global__ __launch_bounds__(256)
void conv3x3_kernel(const __bf16* __restrict__ in,   // [B][C3][NS]
                    const __bf16* __restrict__ wt,   // [9][C3][C3]
                    __bf16* __restrict__ out) {      // [B][C3][NS]
  __shared__ __bf16 sA[128][40];
  __shared__ __bf16 sB[128][40];
  long boff = (long)blockIdx.z * C3 * NS;
  const __bf16* inB = in + boff;
  int n0 = blockIdx.x * 128, m0 = blockIdx.y * 128;
  int tid  = threadIdx.x;
  int lane = tid & 31, wid = tid >> 5;
  int half = lane >> 4, l16 = lane & 15;
  int wm = (wid >> 2) * 64, wn = (wid & 3) * 32;
  v8f acc[4][2] = {};
  int arow = tid >> 1, acb = (tid & 1) * 16;
  int bk = tid >> 3,  bn  = (tid & 7) * 16;
  unsigned ldsA0 = lds_off_of(&sA[arow][acb]);
  unsigned ldsA1 = lds_off_of(&sA[arow][acb + 8]);

  for (int tap = 0; tap < 9; ++tap) {
    int dh = tap / 3 - 1, dw = tap % 3 - 1;
    const __bf16* A = wt + (long)tap * C3 * C3;
    for (int k0 = 0; k0 < C3; k0 += 32) {
      __syncthreads();
      { // A tile via async copy: always in range (M = C3 exact)
        const __bf16* s = A + (long)(m0 + arow) * C3 + k0 + acb;
        async_copy_b128(ldsA0, s);
        async_copy_b128(ldsA1, s + 8);
      }
      { // B tile with (dh,dw) shift + zero pad; 16-wide chunk sits in one image row
        int p0 = n0 + bn;
        int h = p0 / WW, w0 = p0 % WW;
        int h2 = h + dh;
        if ((unsigned)h2 < (unsigned)HH) {
          const __bf16* srow = inB + (long)(k0 + bk) * NS + h2 * WW;
          if (dw == 0) {
            v8bf a0 = *(const v8bf*)(srow + w0);
            v8bf a1 = *(const v8bf*)(srow + w0 + 8);
#pragma unroll
            for (int i = 0; i < 8; ++i) { sB[bn + i][bk] = a0[i]; sB[bn + 8 + i][bk] = a1[i]; }
          } else {
#pragma unroll
            for (int i = 0; i < 16; ++i) {
              int w2 = w0 + i + dw;
              sB[bn + i][bk] = ((unsigned)w2 < (unsigned)WW) ? srow[w2] : (__bf16)0.0f;
            }
          }
        } else {
#pragma unroll
          for (int i = 0; i < 16; ++i) sB[bn + i][bk] = (__bf16)0.0f;
        }
      }
      if (k0 + 32 < C3) {  // prefetch next K tile of A and interior B
        __builtin_prefetch(A + (long)(m0 + arow) * C3 + k0 + 32 + acb, 0, 1);
        __builtin_prefetch(inB + (long)(k0 + 32 + bk) * NS + n0 + bn, 0, 1);
      }
      wait_async0();
      __syncthreads();

      v16bf af[4], bfv[2];
#pragma unroll
      for (int i = 0; i < 4; ++i) {
        const __bf16* p = &sA[wm + i * 16 + l16][0];
        af[i] = mk16(*(const v8bf*)(p + half * 8), *(const v8bf*)(p + 16 + half * 8));
      }
#pragma unroll
      for (int j = 0; j < 2; ++j) {
        const __bf16* p = &sB[wn + j * 16 + l16][0];
        bfv[j] = mk16(*(const v8bf*)(p + half * 8), *(const v8bf*)(p + 16 + half * 8));
      }
#pragma unroll
      for (int i = 0; i < 4; ++i)
#pragma unroll
        for (int j = 0; j < 2; ++j)
          acc[i][j] = WMMA_BF16(af[i], bfv[j], acc[i][j]);
    }
  }

  __bf16* outB = out + boff;
#pragma unroll
  for (int i = 0; i < 4; ++i)
#pragma unroll
    for (int j = 0; j < 2; ++j)
#pragma unroll
      for (int v = 0; v < 8; ++v) {
        int m = m0 + wm + i * 16 + half * 8 + v;
        int n = n0 + wn + j * 16 + l16;
        outB[(long)m * NS + n] = f2bf(acc[i][j][v]);
      }
}

// ---------------------------------------------------------------------------
// Per-channel L2 norm over spatial axis for q and k rows -> 1/max(norm, eps)
// grid.x = BATCH*768 (q rows 0..383, k rows 384..767 of each batch)
// ---------------------------------------------------------------------------
__global__ __launch_bounds__(256)
void norm_kernel(const __bf16* __restrict__ qkv2, float* __restrict__ invn) {
  int r = blockIdx.x;
  int b = r / (2 * DIMC), row = r % (2 * DIMC);
  const __bf16* src = qkv2 + ((long)b * C3 + row) * NS;
  float s = 0.f;
  for (int i = threadIdx.x; i < NS; i += 256) { float v = bf2f(src[i]); s += v * v; }
#pragma unroll
  for (int o = 16; o > 0; o >>= 1) s += __shfl_xor(s, o, 32);
  __shared__ float red[8];
  int wid = threadIdx.x >> 5, lane = threadIdx.x & 31;
  if (lane == 0) red[wid] = s;
  __syncthreads();
  if (threadIdx.x == 0) {
    float t = 0.f;
#pragma unroll
    for (int i = 0; i < 8; ++i) t += red[i];
    invn[r] = 1.0f / fmaxf(sqrtf(t), 1e-12f);
  }
}

// ---------------------------------------------------------------------------
// Attention scores: attn[b,h,c,d] = (q_c . k_d) * invq_c * invk_d * scale_h
// One block per (b,h); 8 waves split K=9216; 3x3 WMMA tiles; LDS f32 reduce.
// ---------------------------------------------------------------------------
__global__ __launch_bounds__(256)
void attn_kernel(const __bf16* __restrict__ qkv2, const float* __restrict__ invn,
                 const float* __restrict__ scale, float* __restrict__ attnS) {
  int z = blockIdx.x;                // b*8 + h
  int b = z >> 3, h = z & 7;
  const __bf16* qb = qkv2 + ((long)b * C3 + h * HD) * NS;
  const __bf16* kb = qkv2 + ((long)b * C3 + DIMC + h * HD) * NS;
  int tid = threadIdx.x, lane = tid & 31, wid = tid >> 5;
  int half = lane >> 4, l16 = lane & 15;

  __shared__ float sAcc[HD * HD];
  for (int i = tid; i < HD * HD; i += 256) sAcc[i] = 0.f;
  __syncthreads();

  v8f acc[3][3] = {};
  int kbase = wid * (NS / 8);        // 1152 per wave
  for (int ks = 0; ks < NS / 8; ks += 32) {
    int k0 = kbase + ks;
    v16bf af[3], bfv[3];
#pragma unroll
    for (int i = 0; i < 3; ++i) {    // A = q rows (M=c)
      const __bf16* p = qb + (long)(i * 16 + l16) * NS + k0;
      af[i] = mk16(*(const v8bf*)(p + half * 8), *(const v8bf*)(p + 16 + half * 8));
    }
#pragma unroll
    for (int j = 0; j < 3; ++j) {    // B = k^T (N=d, K=spatial, half*8+e | 16+half*8+e)
      const __bf16* p = kb + (long)(j * 16 + l16) * NS + k0;
      bfv[j] = mk16(*(const v8bf*)(p + half * 8), *(const v8bf*)(p + 16 + half * 8));
    }
#pragma unroll
    for (int i = 0; i < 3; ++i)
#pragma unroll
      for (int j = 0; j < 3; ++j)
        acc[i][j] = WMMA_BF16(af[i], bfv[j], acc[i][j]);
  }
#pragma unroll
  for (int i = 0; i < 3; ++i)
#pragma unroll
    for (int j = 0; j < 3; ++j)
#pragma unroll
      for (int v = 0; v < 8; ++v) {
        int m = i * 16 + half * 8 + v, n = j * 16 + l16;
        atomicAdd(&sAcc[m * HD + n], acc[i][j][v]);
      }
  __syncthreads();

  float sc = scale[h];
  const float* iq = invn + b * (2 * DIMC) + h * HD;
  const float* ik = invn + b * (2 * DIMC) + DIMC + h * HD;
  for (int idx = tid; idx < HD * HD; idx += 256) {
    int c = idx / HD, d = idx % HD;
    attnS[(long)z * HD * HD + idx] = sAcc[idx] * iq[c] * ik[d] * sc;
  }
}

// ---------------------------------------------------------------------------
// Softmax over d (48) per row; emit bf16 probs zero-padded to 64 cols (K pad).
// One wave32 per row; 8 rows per block.
// ---------------------------------------------------------------------------
__global__ __launch_bounds__(256)
void softmax_kernel(const float* __restrict__ attnS, __bf16* __restrict__ attnP) {
  int row  = blockIdx.x * 8 + (threadIdx.x >> 5);   // 0 .. 4*8*48-1
  int lane = threadIdx.x & 31;
  const float* src = attnS + (long)row * HD;
  float v0 = src[lane];
  float v1 = (lane < 16) ? src[32 + lane] : -3.4e38f;
  float m = fmaxf(v0, v1);
#pragma unroll
  for (int o = 16; o > 0; o >>= 1) m = fmaxf(m, __shfl_xor(m, o, 32));
  float e0 = __expf(v0 - m);
  float e1 = (lane < 16) ? __expf(v1 - m) : 0.f;
  float s = e0 + e1;
#pragma unroll
  for (int o = 16; o > 0; o >>= 1) s += __shfl_xor(s, o, 32);
  float inv = 1.f / s;
  __bf16* dst = attnP + (long)row * 64;
  dst[lane]      = f2bf(e0 * inv);
  dst[32 + lane] = (lane < 16) ? f2bf(e1 * inv) : (__bf16)0.0f;   // cols 48..63 = 0
}

// ---------------------------------------------------------------------------
// Host launch
// ---------------------------------------------------------------------------
extern "C" void kernel_launch(void* const* d_in, const int* in_sizes, int n_in,
                              void* d_out, int out_size, void* d_ws, size_t ws_size,
                              hipStream_t stream) {
  (void)in_sizes; (void)n_in; (void)out_size; (void)ws_size;
  const float* x      = (const float*)d_in[0];   // [4,384,96,96]
  const float* qkv_w  = (const float*)d_in[1];   // [1152,384,1,1]
  const float* enh_w  = (const float*)d_in[2];   // [1152,1152,3,3]
  const float* out_w  = (const float*)d_in[3];   // [384,384,1,1]
  const float* scale  = (const float*)d_in[4];   // [8,1,1]
  float*       out    = (float*)d_out;           // [4,384,96,96]

  char* ws = (char*)d_ws;
  size_t off = 0;
  auto alloc = [&](size_t bytes) -> void* {
    void* p = ws + off;
    off = (off + bytes + 255) & ~(size_t)255;
    return p;
  };
  const size_t nX = (size_t)BATCH * DIMC * NS;     // 14,155,776
  __bf16* xb     = (__bf16*)alloc(nX * 2);
  __bf16* wqkv_b = (__bf16*)alloc((size_t)C3 * DIMC * 2);
  __bf16* wout_b = (__bf16*)alloc((size_t)DIMC * DIMC * 2);
  __bf16* wenh_t = (__bf16*)alloc((size_t)9 * C3 * C3 * 2);
  __bf16* qkv1_b = (__bf16*)alloc((size_t)BATCH * C3 * NS * 2);
  __bf16* qkv2_b = (__bf16*)alloc(((size_t)BATCH * C3 + 16) * NS * 2);  // +16 rows slack for K-pad reads
  float*  invn   = (float*)alloc((size_t)BATCH * 2 * DIMC * 4);
  float*  attnS  = (float*)alloc((size_t)BATCH * HEADS * HD * HD * 4);
  __bf16* attnP  = (__bf16*)alloc((size_t)BATCH * HEADS * HD * 64 * 2);
  __bf16* av_b   = (__bf16*)alloc((size_t)BATCH * DIMC * NS * 2);

  // 1) bf16 conversions / weight repack
  cvt_bf16_kernel<<<(int)((nX + 255) / 256), 256, 0, stream>>>(x, xb, (int)nX);
  cvt_bf16_kernel<<<(C3 * DIMC + 255) / 256, 256, 0, stream>>>(qkv_w, wqkv_b, C3 * DIMC);
  cvt_bf16_kernel<<<(DIMC * DIMC + 255) / 256, 256, 0, stream>>>(out_w, wout_b, DIMC * DIMC);
  repack_enh_kernel<<<(9 * C3 * C3 + 255) / 256, 256, 0, stream>>>(enh_w, wenh_t);

  // 2) qkv 1x1 conv: per batch [1152,384] x [384,9216]
  gemm_kernel<true><<<dim3(NS / 128, C3 / 128, BATCH), 256, 0, stream>>>(
      wqkv_b, xb, qkv1_b, C3, NS, DIMC, 1,
      0L, 0L, (long)DIMC * NS, 0L, (long)C3 * NS, 0L);

  // 3) enhancement 3x3 conv (dominant: ~881 GFLOP)
  conv3x3_kernel<<<dim3(NS / 128, C3 / 128, BATCH), 256, 0, stream>>>(qkv1_b, wenh_t, qkv2_b);

  // 4) q/k L2 norms over spatial
  norm_kernel<<<BATCH * 2 * DIMC, 256, 0, stream>>>(qkv2_b, invn);

  // 5) attention scores + scaling
  attn_kernel<<<BATCH * HEADS, 256, 0, stream>>>(qkv2_b, invn, scale, attnS);

  // 6) softmax -> bf16 probs padded [48,64]
  softmax_kernel<<<BATCH * HEADS * HD / 8, 256, 0, stream>>>(attnS, attnP);

  // 7) out = attn @ v : per (b,h) [48,64] x [64,9216] (cols 48..63 of A are 0)
  gemm_kernel<true><<<dim3(NS / 128, 1, BATCH * HEADS), 256, 0, stream>>>(
      attnP, qkv2_b + (size_t)2 * DIMC * NS, av_b, HD, NS, 64, HEADS,
      (long)HEADS * HD * 64, (long)HD * 64,
      (long)C3 * NS, (long)HD * NS,
      (long)DIMC * NS, (long)HD * NS);

  // 8) output 1x1 conv: per batch [384,384] x [384,9216] -> f32 d_out
  gemm_kernel<false><<<dim3(NS / 128, DIMC / 128, BATCH), 256, 0, stream>>>(
      wout_b, av_b, out, DIMC, NS, DIMC, 1,
      0L, 0L, (long)DIMC * NS, 0L, (long)DIMC * NS, 0L);
}
